// ResGNet_81612968558773
// MI455X (gfx1250) — matrix operand order repaired and checked by reading, not compiled, not run on hardware
//
#include <hip/hip_runtime.h>
#include <cstdint>

#define NN   10000
#define EE   160000
#define CIN  64
#define COUT 128
#define KS   5
#define KTOT 125      // 5^3
#define K1A  (KTOT*CIN)    // 8000
#define K2A  (KTOT*COUT)   // 16000

typedef __attribute__((ext_vector_type(8)))  float        v8f;
typedef __attribute__((ext_vector_type(16))) __bf16       v16bf;
typedef __attribute__((ext_vector_type(8)))  unsigned int v8u;

// ---------- helpers ----------
// Used only in the (cold) weight-pack kernel; hot path uses native (__bf16) casts.
__device__ __forceinline__ unsigned short f2bf_bits(float f) {
  union { float f; unsigned u; } x; x.f = f;
  unsigned r = x.u + 0x7FFFu + ((x.u >> 16) & 1u);   // round-to-nearest-even
  return (unsigned short)(r >> 16);
}

// ---------- zero workspace (accumulators + degree + stats) ----------
__global__ void zero_k(float4* __restrict__ p, size_t n4) {
  size_t t = (size_t)blockIdx.x * blockDim.x + threadIdx.x;
  size_t stride = (size_t)gridDim.x * blockDim.x;
  float4 z = make_float4(0.f, 0.f, 0.f, 0.f);
  for (size_t i = t; i < n4; i += stride) p[i] = z;
}

// ---------- pack W [K,128] f32 -> bf16 WMMA B-fragment layout ----------
// Packed dword index: (((kt*8 + nt)*32 + lane)*8 + j)
// lane<16 -> K base 0, lane>=16 -> K base 8; dword j holds K=kb+2j+(j>=4?8:0) (lo), +1 (hi); N = nt*16 + (lane&15)
__global__ void pack_b(const float* __restrict__ W, unsigned int* __restrict__ P, int K) {
  int t = blockIdx.x * blockDim.x + threadIdx.x;
  int total = (K >> 5) * 2048;                 // (K/32)*8*32*8
  if (t >= total) return;
  int j    = t & 7;
  int lane = (t >> 3) & 31;
  int nt   = (t >> 8) & 7;
  int kt   = t >> 11;
  int kb   = (lane >> 4) << 3;                 // 0 or 8
  int k0   = kt * 32 + kb + 2 * j + ((j >> 2) << 3);
  int n    = nt * 16 + (lane & 15);
  unsigned lo = f2bf_bits(W[(size_t)k0 * COUT + n]);
  unsigned hi = f2bf_bits(W[(size_t)(k0 + 1) * COUT + n]);
  P[t] = (hi << 16) | lo;
}

// ---------- degree ----------
__global__ void degree_k(const int* __restrict__ dst, float* __restrict__ deg) {
  int e = blockIdx.x * blockDim.x + threadIdx.x;
  if (e < EE) unsafeAtomicAdd(&deg[dst[e]], 1.0f);
}

// ---------- spline scatter: acc[dst, idx, c] += w * feat[src, c] (8 corners) ----------
__global__ void scatter_spline(const float* __restrict__ feat,
                               const int* __restrict__ src,
                               const int* __restrict__ dst,
                               const float* __restrict__ attr,
                               float* __restrict__ acc, int logC) {
  int t = blockIdx.x * blockDim.x + threadIdx.x;
  int e = t >> logC;
  if (e >= EE) return;
  int C = 1 << logC;
  int c = t & (C - 1);
  int s = src[e], d = dst[e];
  float xs = feat[(size_t)s * C + c];
  float fr[3]; int bi[3];
#pragma unroll
  for (int dd = 0; dd < 3; ++dd) {
    float v = attr[(size_t)e * 3 + dd] * (float)(KS - 1);
    float b = floorf(v);
    fr[dd] = v - b;
    bi[dd] = (int)b;
  }
  float* base = acc + ((size_t)d * KTOT) * C + c;
#pragma unroll
  for (int combo = 0; combo < 8; ++combo) {
    float w = 1.0f; int idx = 0;
#pragma unroll
    for (int dd = 0; dd < 3; ++dd) {
      int bit = (combo >> dd) & 1;
      w *= bit ? fr[dd] : (1.0f - fr[dd]);
      int ii = bi[dd] + bit;
      ii = ii < 0 ? 0 : (ii > KS - 1 ? KS - 1 : ii);
      idx = idx * KS + ii;
    }
    unsafeAtomicAdd(base + (size_t)idx * C, w * xs);
  }
}

// ---------- K=1 shortcut scatter (basis weight == 1, single bin) ----------
__global__ void scatter_mean(const float* __restrict__ feat,
                             const int* __restrict__ src,
                             const int* __restrict__ dst,
                             float* __restrict__ acc) {
  int t = blockIdx.x * blockDim.x + threadIdx.x;
  int e = t >> 6;
  if (e >= EE) return;
  int c = t & 63;
  unsafeAtomicAdd(&acc[(size_t)dst[e] * CIN + c], feat[(size_t)src[e] * CIN + c]);
}

// ---------- WMMA GEMM segment: c += bf16(A_row) x B_packed (no scaling here) ----------
__device__ __forceinline__ void gemm_seg(const float* __restrict__ arow, int ktiles,
                                         const unsigned int* __restrict__ BP,
                                         int nt, int lane, int kb, v8f& c) {
  for (int kt = 0; kt < ktiles; ++kt) {
    const float* p = arow + kt * 32 + kb;
    __builtin_prefetch(p + 256, 0, 1);                 // global_prefetch ahead in A row
    float4 a0 = *reinterpret_cast<const float4*>(p);
    float4 a1 = *reinterpret_cast<const float4*>(p + 4);
    float4 a2 = *reinterpret_cast<const float4*>(p + 16);
    float4 a3 = *reinterpret_cast<const float4*>(p + 20);
    v8u braw = *reinterpret_cast<const v8u*>(BP + (((size_t)kt * 8 + nt) * 32 + lane) * 8);
    // native fptrunc -> v_cvt_pk_bf16_f32 pairs (cheap VALU, co-executes with WMMA)
    v16bf a;
    a[0]  = (__bf16)a0.x; a[1]  = (__bf16)a0.y; a[2]  = (__bf16)a0.z; a[3]  = (__bf16)a0.w;
    a[4]  = (__bf16)a1.x; a[5]  = (__bf16)a1.y; a[6]  = (__bf16)a1.z; a[7]  = (__bf16)a1.w;
    a[8]  = (__bf16)a2.x; a[9]  = (__bf16)a2.y; a[10] = (__bf16)a2.z; a[11] = (__bf16)a2.w;
    a[12] = (__bf16)a3.x; a[13] = (__bf16)a3.y; a[14] = (__bf16)a3.z; a[15] = (__bf16)a3.w;
    v16bf b = __builtin_bit_cast(v16bf, braw);
    c = __builtin_amdgcn_wmma_f32_16x16x32_bf16(false, a, false, b, (short)0, c, false, false);
  }
}

// ---------- fused SplineConv GEMM: out = (acc@W)/deg + root@Wr + bias ----------
// block = 256 threads = 8 waves; block handles one 16-row M tile; wave w -> N tile w.
// Mean-divide folded into the epilogue: diag(1/deg)*(acc@W) == (diag(1/deg)*acc)@W.
__global__ void __launch_bounds__(256)
spline_gemm(const float* __restrict__ Aacc, int Kacc,
            const float* __restrict__ Aroot, int Kroot,
            const unsigned int* __restrict__ BPacc,
            const unsigned int* __restrict__ BProot,
            const float* __restrict__ deg,
            const float* __restrict__ bias,
            float* __restrict__ out) {
  int lane  = threadIdx.x & 31;
  int nt    = threadIdx.x >> 5;                 // 0..7
  int mtile = blockIdx.x;                       // 0..624 (625*16 == 10000)
  int mrow  = mtile * 16 + (lane & 15);         // A row for this lane (both halves)
  int kb    = (lane >> 4) << 3;                 // A K-offset: 0 or 8 (ISA 16-bit A layout)
  v8f c1 = {};                                  // acc segment (needs /deg)
  v8f c2 = {};                                  // root segment (no scale)
  gemm_seg(Aacc  + (size_t)mrow * Kacc,  Kacc  >> 5, BPacc,  nt, lane, kb, c1);
  gemm_seg(Aroot + (size_t)mrow * Kroot, Kroot >> 5, BProot, nt, lane, kb, c2);
  int n  = nt * 16 + (lane & 15);
  float bv = bias[n];
  int rbase = mtile * 16 + ((lane >> 4) << 3);  // C layout: VGPR r -> M = r (+8 for hi half)
#pragma unroll
  for (int r = 0; r < 8; ++r) {
    float sc = 1.0f / fmaxf(deg[rbase + r], 1.0f);
    out[(size_t)(rbase + r) * COUT + n] = fmaf(c1[r], sc, c2[r] + bv);
  }
}

// ---------- batch-norm statistics (sum, sumsq per channel) ----------
__global__ void bn_stats(const float* __restrict__ h, float* __restrict__ st) {
  int c = threadIdx.x;                          // 128 threads = channels
  float s = 0.f, q = 0.f;
  for (int r = blockIdx.x; r < NN; r += gridDim.x) {
    float v = h[(size_t)r * COUT + c];
    s += v; q += v * v;
  }
  unsafeAtomicAdd(&st[c], s);
  unsafeAtomicAdd(&st[COUT + c], q);
}

__device__ __forceinline__ float bn_apply(float v, const float* st, const float* g,
                                          const float* b, int c) {
  float mu  = st[c] * (1.0f / NN);
  float var = st[COUT + c] * (1.0f / NN) - mu * mu;
  return (v - mu) * __frsqrt_rn(var + 1e-5f) * g[c] + b[c];
}

__global__ void bn_elu(const float* __restrict__ h, const float* __restrict__ st,
                       const float* __restrict__ g, const float* __restrict__ b,
                       float* __restrict__ out) {
  int t = blockIdx.x * blockDim.x + threadIdx.x;
  if (t >= NN * COUT) return;
  float y = bn_apply(h[t], st, g, b, t & (COUT - 1));
  out[t] = y > 0.f ? y : (__expf(y) - 1.f);
}

__global__ void final_fuse(const float* __restrict__ h2, const float* __restrict__ s,
                           const float* __restrict__ st2, const float* __restrict__ stsc,
                           const float* __restrict__ g2, const float* __restrict__ b2,
                           const float* __restrict__ gsc, const float* __restrict__ bsc,
                           float* __restrict__ out) {
  int t = blockIdx.x * blockDim.x + threadIdx.x;
  if (t >= NN * COUT) return;
  int c = t & (COUT - 1);
  float y = bn_apply(h2[t], st2, g2, b2, c) + bn_apply(s[t], stsc, gsc, bsc, c);
  out[t] = y > 0.f ? y : (__expf(y) - 1.f);
}

// ---------- launch ----------
extern "C" void kernel_launch(void* const* d_in, const int* in_sizes, int n_in,
                              void* d_out, int out_size, void* d_ws, size_t ws_size,
                              hipStream_t stream) {
  const float* x    = (const float*)d_in[0];
  const int*   ei   = (const int*)d_in[1];
  const float* attr = (const float*)d_in[2];
  const float* W1   = (const float*)d_in[3];
  const float* Wr1  = (const float*)d_in[4];
  const float* b1   = (const float*)d_in[5];
  const float* g1   = (const float*)d_in[6];
  const float* be1  = (const float*)d_in[7];
  const float* W2   = (const float*)d_in[8];
  const float* Wr2  = (const float*)d_in[9];
  const float* b2   = (const float*)d_in[10];
  const float* g2   = (const float*)d_in[11];
  const float* be2  = (const float*)d_in[12];
  const float* Wsc  = (const float*)d_in[13];
  const float* Wrsc = (const float*)d_in[14];
  const float* bsc  = (const float*)d_in[15];
  const float* gsc  = (const float*)d_in[16];
  const float* besc = (const float*)d_in[17];
  const int* src = ei;
  const int* dst = ei + EE;
  float* out = (float*)d_out;

  // workspace layout (floats)
  float* ws    = (float*)d_ws;
  float* acc1  = ws;                                  // NN*125*64
  float* acc2  = acc1  + (size_t)NN * KTOT * CIN;     // NN*125*128
  float* accsc = acc2  + (size_t)NN * KTOT * COUT;    // NN*64
  float* deg   = accsc + (size_t)NN * CIN;            // NN
  float* st1   = deg   + NN;                          // 256
  float* st2   = st1   + 2 * COUT;
  float* stsc  = st2   + 2 * COUT;
  size_t zspan = (size_t)NN * KTOT * CIN + (size_t)NN * KTOT * COUT +
                 (size_t)NN * CIN + NN + 6 * COUT;    // 240,650,768 floats
  float* h1pre = stsc  + 2 * COUT;                    // NN*128 each
  float* h1    = h1pre + (size_t)NN * COUT;
  float* h2pre = h1    + (size_t)NN * COUT;
  float* spre  = h2pre + (size_t)NN * COUT;
  unsigned int* wp1  = (unsigned int*)(spre + (size_t)NN * COUT);
  unsigned int* wr1  = wp1  + (size_t)K1A  * 64;      // K*64 dwords each
  unsigned int* wp2  = wr1  + (size_t)CIN  * 64;
  unsigned int* wr2  = wp2  + (size_t)K2A  * 64;
  unsigned int* wpsc = wr2  + (size_t)COUT * 64;
  unsigned int* wrsc = wpsc + (size_t)CIN  * 64;

  // 1) zero accumulators / degree / stats
  zero_k<<<4096, 256, 0, stream>>>((float4*)ws, zspan / 4);

  // 2) pack weights into bf16 WMMA B-fragments
  pack_b<<<(K1A * 64 + 255) / 256, 256, 0, stream>>>(W1, wp1, K1A);
  pack_b<<<(CIN * 64 + 255) / 256, 256, 0, stream>>>(Wr1, wr1, CIN);
  pack_b<<<(K2A * 64 + 255) / 256, 256, 0, stream>>>(W2, wp2, K2A);
  pack_b<<<(COUT * 64 + 255) / 256, 256, 0, stream>>>(Wr2, wr2, COUT);
  pack_b<<<(CIN * 64 + 255) / 256, 256, 0, stream>>>(Wsc, wpsc, CIN);
  pack_b<<<(CIN * 64 + 255) / 256, 256, 0, stream>>>(Wrsc, wrsc, CIN);

  // 3) degree + scatters feeding conv1 and shortcut
  degree_k<<<(EE + 255) / 256, 256, 0, stream>>>(dst, deg);
  scatter_spline<<<EE * 64 / 256, 256, 0, stream>>>(x, src, dst, attr, acc1, 6);
  scatter_mean<<<EE * 64 / 256, 256, 0, stream>>>(x, src, dst, accsc);

  // 4) conv1 GEMM -> bn1 -> elu
  spline_gemm<<<NN / 16, 256, 0, stream>>>(acc1, K1A, x, CIN, wp1, wr1, deg, b1, h1pre);
  bn_stats<<<256, 128, 0, stream>>>(h1pre, st1);
  bn_elu<<<NN * COUT / 256, 256, 0, stream>>>(h1pre, st1, g1, be1, h1);

  // 5) conv2 scatter + GEMM -> bn2 stats
  scatter_spline<<<EE * 128 / 256, 256, 0, stream>>>(h1, src, dst, attr, acc2, 7);
  spline_gemm<<<NN / 16, 256, 0, stream>>>(acc2, K2A, h1, COUT, wp2, wr2, deg, b2, h2pre);
  bn_stats<<<256, 128, 0, stream>>>(h2pre, st2);

  // 6) shortcut GEMM -> bnsc stats
  spline_gemm<<<NN / 16, 256, 0, stream>>>(accsc, CIN, x, CIN, wpsc, wrsc, deg, bsc, spre);
  bn_stats<<<256, 128, 0, stream>>>(spre, stsc);

  // 7) out = elu(bn2(h2pre) + bnsc(spre))
  final_fuse<<<NN * COUT / 256, 256, 0, stream>>>(h2pre, spre, st2, stsc,
                                                  g2, be2, gsc, besc, out);
}
